// EGNN4KLIFF_73675868995974
// MI455X (gfx1250) — matrix operand order repaired and checked by reading, not compiled, not run on hardware
//
#include <hip/hip_runtime.h>
#include <hip/hip_bf16.h>

// ---------------------------------------------------------------------------
// EGNN for KLIFF on gfx1250: edge/node MLPs as wave32 WMMA f16->f32 GEMMs.
// Weights pre-converted to f16 once per launch; B fragments are pure b128
// loads, A fragments are pure ds_load_b128 pairs. Fast activations via
// v_rcp_f32 (no IEEE div chains).
// ---------------------------------------------------------------------------

typedef __attribute__((ext_vector_type(16))) _Float16 v16h;
typedef __attribute__((ext_vector_type(8)))  _Float16 v8h;
typedef __attribute__((ext_vector_type(4)))  _Float16 v4h;
typedef __attribute__((ext_vector_type(8)))  float    v8f;

#define H     128
#define WAVES 4
#define TPB   (WAVES * 32)

// silu(x) = x * sigmoid(x); fast reciprocal (v_rcp_f32), no div chains
__device__ __forceinline__ float silu_f(float x) {
  return x * __builtin_amdgcn_rcpf(1.0f + __expf(-x));
}
// tanh(x) = (1 - e) / (1 + e), e = exp(-2x)
__device__ __forceinline__ float tanh_f(float x) {
  float e = __expf(-2.0f * x);
  return (1.0f - e) * __builtin_amdgcn_rcpf(1.0f + e);
}

// A fragment (16x32 f16, ISA layout): lane L, row m=L&15, half hs=L>>4:
// halves 0..7  = A[m][kb + hs*8 .. +7], halves 8..15 = A[m][kb + 16 + hs*8 .. +7]
// Both runs 16B-aligned -> two ds_load_b128.
__device__ __forceinline__ v16h a_frag_lds(const _Float16* A, int ldA, int kb, int lane) {
  int m = lane & 15, hs = lane >> 4;
  const _Float16* base = A + m * ldA + kb + hs * 8;
  v8h lo = *(const v8h*)base;
  v8h hi = *(const v8h*)(base + 16);
  return __builtin_shufflevector(lo, hi, 0, 1, 2, 3, 4, 5, 6, 7,
                                 8, 9, 10, 11, 12, 13, 14, 15);
}

// B fragment (32x16 f16, ISA layout): lane L, col n=L&15, ks=L>>4:
// halves 0..7 = W[nb+n][kb + ks*8 .. +7], halves 8..15 = W[nb+n][kb + 16 + ks*8 .. +7]
// (B[k][n] = W[n][k]) -- f16 weights, 16B-aligned runs -> two global_load_b128.
__device__ __forceinline__ v16h b_frag_h(const _Float16* __restrict__ W, int ldW,
                                         int nb, int kb, int lane) {
  int n = lane & 15, ks = lane >> 4;
  const v8h* p = (const v8h*)(W + (size_t)(nb + n) * ldW + kb + ks * 8);
  v8h lo = p[0], hi = p[2];
  return __builtin_shufflevector(lo, hi, 0, 1, 2, 3, 4, 5, 6, 7,
                                 8, 9, 10, 11, 12, 13, 14, 15);
}

#define WMMA_F16(acc, a, b) \
  __builtin_amdgcn_wmma_f32_16x16x32_f16(false, (a), false, (b), (short)0, (acc), false, false)

// ---------------------------------------------------------------------------
// weight f32 -> f16 conversion (once per launch, reused by every tile)
// ---------------------------------------------------------------------------
__global__ void cvt_w_kernel(const float* __restrict__ src, _Float16* __restrict__ dst,
                             int rows, int src_ld, int dst_ld) {
  int idx = blockIdx.x * blockDim.x + threadIdx.x;
  if (idx >= rows * dst_ld) return;
  int row = idx / dst_ld, col = idx - row * dst_ld;
  dst[idx] = (_Float16)src[(size_t)row * src_ld + col];
}

// ---------------------------------------------------------------------------
// max(x) for positive floats via uint atomicMax
// ---------------------------------------------------------------------------
__global__ void xmax_kernel(const float* __restrict__ x, unsigned* __restrict__ xmax, int n) {
  int i = blockIdx.x * blockDim.x + threadIdx.x;
  float v = (i < n) ? x[i] : 0.0f;
#pragma unroll
  for (int off = 16; off > 0; off >>= 1) v = fmaxf(v, __shfl_xor(v, off, 32));
  if ((threadIdx.x & 31) == 0) atomicMax(xmax, __float_as_uint(v));
}

// h = [1, t, t^2] @ Wemb^T + bemb,  t = x/max(x)
__global__ void embed_kernel(const float* __restrict__ x, const unsigned* __restrict__ xmax,
                             const float* __restrict__ W, const float* __restrict__ b,
                             float* __restrict__ hout, int nN) {
  int idx = blockIdx.x * blockDim.x + threadIdx.x;
  if (idx >= nN * H) return;
  int node = idx >> 7, f = idx & (H - 1);
  float t = x[node] / __uint_as_float(xmax[0]);
  hout[idx] = b[f] + W[f * 3 + 0] + W[f * 3 + 1] * t + W[f * 3 + 2] * t * t;
}

__global__ void radd_kernel(const float* __restrict__ rin, const float* __restrict__ racc,
                            float* __restrict__ rout, int n3) {
  int i = blockIdx.x * blockDim.x + threadIdx.x;
  if (i < n3) rout[i] = rin[i] + racc[i];
}

// ---------------------------------------------------------------------------
// EGCL edge kernel: mij = phi_e([h_dst | h_src | |dr|^2]); s = phi_r(mij);
// scatter mij -> m[dst], dr*s -> racc[dst].  One wave per 16-edge tile.
// ---------------------------------------------------------------------------
__global__ __launch_bounds__(TPB)
void egcl_edge_kernel(const float* __restrict__ h, const float* __restrict__ r,
                      const int* __restrict__ eidx,            // [2,E] row0=src row1=dst
                      const float* __restrict__ We1f,          // f32 [H,257] (col 256 only)
                      const _Float16* __restrict__ We1,        // f16 [H,256]
                      const float* __restrict__ be1,
                      const _Float16* __restrict__ We2, const float* __restrict__ be2,
                      const _Float16* __restrict__ Wr1, const float* __restrict__ br1,
                      const float* __restrict__ wr2, const float* __restrict__ br2,
                      float* __restrict__ m_acc, float* __restrict__ r_acc, int nE) {
  __shared__ _Float16 A1[WAVES][16][2 * H];  // [h_dst|h_src], later reused for mij
  __shared__ _Float16 A2[WAVES][16][H];
  __shared__ float    nrm[WAVES][16];
  __shared__ float    drb[WAVES][16][3];
  __shared__ int      dstb[WAVES][16];

  int lane = threadIdx.x & 31, wv = threadIdx.x >> 5;
  int tile = blockIdx.x * WAVES + wv;
  int e0 = tile * 16;
  if (e0 >= nE) return;
  const int* srcA = eidx;
  const int* dstA = eidx + nE;
  int mrow = lane & 15, hs = lane >> 4;

  // ---- stage: geometry + gathered node features (f32 -> f16 into LDS) ----
  int e = e0 + mrow; if (e >= nE) e = nE - 1;
  int s_i = srcA[e], d_i = dstA[e];
  if (hs == 0) {
    float dx = r[s_i * 3 + 0] - r[d_i * 3 + 0];
    float dy = r[s_i * 3 + 1] - r[d_i * 3 + 1];
    float dz = r[s_i * 3 + 2] - r[d_i * 3 + 2];
    nrm[wv][mrow] = dx * dx + dy * dy + dz * dz;
    drb[wv][mrow][0] = dx; drb[wv][mrow][1] = dy; drb[wv][mrow][2] = dz;
    dstb[wv][mrow] = d_i;
  }
  {
    int node = (hs == 0) ? d_i : s_i;
    const float4* hp = (const float4*)(h + (size_t)node * H);
    v4h* ap = (v4h*)(&A1[wv][mrow][hs * H]);
#pragma unroll
    for (int j = 0; j < H / 4; ++j) {
      float4 v = hp[j];
      v4h o; o[0] = (_Float16)v.x; o[1] = (_Float16)v.y; o[2] = (_Float16)v.z; o[3] = (_Float16)v.w;
      ap[j] = o;
    }
  }
  asm volatile("s_wait_dscnt 0" ::: "memory");

  // ---- G1: silu(A1[16x256] @ We1^T + be1 + nrm*We1f[:,256]) -> A2 ----
  for (int nt = 0; nt < 8; ++nt) {
    int n = nt * 16 + (lane & 15);
    float bias = be1[n];
    float wlast = We1f[(size_t)n * 257 + 256];
    v8f acc;
#pragma unroll
    for (int v = 0; v < 8; ++v) acc[v] = bias + nrm[wv][v + 8 * hs] * wlast;
#pragma unroll
    for (int kb = 0; kb < 2 * H; kb += 32) {
      v16h a = a_frag_lds(&A1[wv][0][0], 2 * H, kb, lane);
      v16h b = b_frag_h(We1, 2 * H, nt * 16, kb, lane);
      acc = WMMA_F16(acc, a, b);
    }
#pragma unroll
    for (int v = 0; v < 8; ++v) A2[wv][v + 8 * hs][n] = (_Float16)silu_f(acc[v]);
  }
  asm volatile("s_wait_dscnt 0" ::: "memory");

  // ---- G2: mij = silu(A2 @ We2^T + be2); scatter to m[dst]; stash f16 copy ----
  for (int nt = 0; nt < 8; ++nt) {
    int n = nt * 16 + (lane & 15);
    v8f acc;
#pragma unroll
    for (int v = 0; v < 8; ++v) acc[v] = be2[n];
#pragma unroll
    for (int kb = 0; kb < H; kb += 32) {
      v16h a = a_frag_lds(&A2[wv][0][0], H, kb, lane);
      v16h b = b_frag_h(We2, H, nt * 16, kb, lane);
      acc = WMMA_F16(acc, a, b);
    }
#pragma unroll
    for (int v = 0; v < 8; ++v) {
      float mv = silu_f(acc[v]);
      int row = v + 8 * hs;
      A1[wv][row][n] = (_Float16)mv;  // reuse A1 (ld 2H) as mij tile
      int e2 = e0 + row;
      if (e2 < nE) atomicAdd(&m_acc[(size_t)dstb[wv][row] * H + n], mv);
    }
  }
  asm volatile("s_wait_dscnt 0" ::: "memory");

  // ---- G3: t = silu(mij @ Wr1^T + br1); s = tanh(t . wr2 + br2); delr atomics ----
  float part[8];
#pragma unroll
  for (int v = 0; v < 8; ++v) part[v] = 0.0f;
  for (int nt = 0; nt < 8; ++nt) {
    int n = nt * 16 + (lane & 15);
    float w2 = wr2[n];
    v8f acc;
#pragma unroll
    for (int v = 0; v < 8; ++v) acc[v] = br1[n];
#pragma unroll
    for (int kb = 0; kb < H; kb += 32) {
      v16h a = a_frag_lds(&A1[wv][0][0], 2 * H, kb, lane);
      v16h b = b_frag_h(Wr1, H, nt * 16, kb, lane);
      acc = WMMA_F16(acc, a, b);
    }
#pragma unroll
    for (int v = 0; v < 8; ++v) part[v] += silu_f(acc[v]) * w2;
  }
#pragma unroll
  for (int off = 1; off < 16; off <<= 1) {
#pragma unroll
    for (int v = 0; v < 8; ++v) part[v] += __shfl_xor(part[v], off, 32);
  }
  float br2v = br2[0];
  int q = lane & 15;
  if (q < 3) {
#pragma unroll
    for (int v = 0; v < 8; ++v) {
      int row = v + 8 * hs, e2 = e0 + row;
      if (e2 < nE) {
        float s = tanh_f(part[v] + br2v);
        atomicAdd(&r_acc[(size_t)dstb[wv][row] * 3 + q], drb[wv][row][q] * s);
      }
    }
  }
}

// ---------------------------------------------------------------------------
// EGCL node kernel: h' = phi_h([h|m]); blend by mask(sum|m|>0); restore rows
// where sum|h1|==0.  One wave per 16-node tile.
// ---------------------------------------------------------------------------
__global__ __launch_bounds__(TPB)
void egcl_node_kernel(const float* __restrict__ h, const float* __restrict__ m_in,
                      const _Float16* __restrict__ Wh1, const float* __restrict__ bh1,
                      const _Float16* __restrict__ Wh2, const float* __restrict__ bh2,
                      float* __restrict__ h_out, int nN) {
  __shared__ _Float16 A1[WAVES][16][2 * H];
  __shared__ _Float16 A2[WAVES][16][H];
  __shared__ float    msum[WAVES][16];

  int lane = threadIdx.x & 31, wv = threadIdx.x >> 5;
  int tile = blockIdx.x * WAVES + wv;
  int n0 = tile * 16;
  if (n0 >= nN) return;
  int mrow = lane & 15, hs = lane >> 4;
  int node = n0 + mrow; if (node >= nN) node = nN - 1;

  {
    const float* src = (hs == 0) ? (h + (size_t)node * H) : (m_in + (size_t)node * H);
    const float4* sp = (const float4*)src;
    v4h* ap = (v4h*)(&A1[wv][mrow][hs * H]);
    float asum = 0.0f;
#pragma unroll
    for (int j = 0; j < H / 4; ++j) {
      float4 v = sp[j];
      asum += fabsf(v.x) + fabsf(v.y) + fabsf(v.z) + fabsf(v.w);
      v4h o; o[0] = (_Float16)v.x; o[1] = (_Float16)v.y; o[2] = (_Float16)v.z; o[3] = (_Float16)v.w;
      ap[j] = o;
    }
    if (hs == 1) msum[wv][mrow] = asum;  // sum|m| per row
  }
  asm volatile("s_wait_dscnt 0" ::: "memory");

  // G1: silu(A1[16x256] @ Wh1^T + bh1) -> A2
  for (int nt = 0; nt < 8; ++nt) {
    int n = nt * 16 + (lane & 15);
    v8f acc;
#pragma unroll
    for (int v = 0; v < 8; ++v) acc[v] = bh1[n];
#pragma unroll
    for (int kb = 0; kb < 2 * H; kb += 32) {
      v16h a = a_frag_lds(&A1[wv][0][0], 2 * H, kb, lane);
      v16h b = b_frag_h(Wh1, 2 * H, nt * 16, kb, lane);
      acc = WMMA_F16(acc, a, b);
    }
#pragma unroll
    for (int v = 0; v < 8; ++v) A2[wv][v + 8 * hs][n] = (_Float16)silu_f(acc[v]);
  }
  asm volatile("s_wait_dscnt 0" ::: "memory");

  // G2: h' = A2 @ Wh2^T + bh2; blend with h by mask; track sum|h1| per row
  float rowabs[8];
#pragma unroll
  for (int v = 0; v < 8; ++v) rowabs[v] = 0.0f;
  for (int nt = 0; nt < 8; ++nt) {
    int n = nt * 16 + (lane & 15);
    v8f acc;
#pragma unroll
    for (int v = 0; v < 8; ++v) acc[v] = bh2[n];
#pragma unroll
    for (int kb = 0; kb < H; kb += 32) {
      v16h a = a_frag_lds(&A2[wv][0][0], H, kb, lane);
      v16h b = b_frag_h(Wh2, H, nt * 16, kb, lane);
      acc = WMMA_F16(acc, a, b);
    }
#pragma unroll
    for (int v = 0; v < 8; ++v) {
      int row = v + 8 * hs, nv = n0 + row;
      if (nv < nN) {
        float hv = h[(size_t)nv * H + n];
        float h1 = (msum[wv][row] > 0.0f) ? acc[v] : hv;
        rowabs[v] += fabsf(h1);
        h_out[(size_t)nv * H + n] = h1;
      }
    }
  }
#pragma unroll
  for (int off = 1; off < 16; off <<= 1) {
#pragma unroll
    for (int v = 0; v < 8; ++v) rowabs[v] += __shfl_xor(rowabs[v], off, 32);
  }
  // rare restore pass: rows where h1 is identically zero get h added back
#pragma unroll
  for (int v = 0; v < 8; ++v) {
    int row = v + 8 * hs, nv = n0 + row;
    if (nv < nN && rowabs[v] == 0.0f) {
      for (int nt = 0; nt < 8; ++nt) {
        int n = nt * 16 + (lane & 15);
        h_out[(size_t)nv * H + n] += h[(size_t)nv * H + n];
      }
    }
  }
}

// ---------------------------------------------------------------------------
// Final MLP: E_local = w3 . silu(W2 @ silu(W1 @ h + b1) + b2) + b3;
// atomicAdd into out[0] where contributions == 0  (Eg[::2] -> segment 0).
// ---------------------------------------------------------------------------
__global__ __launch_bounds__(TPB)
void mlp_kernel(const float* __restrict__ h,
                const _Float16* __restrict__ Wm1, const float* __restrict__ bm1,
                const _Float16* __restrict__ Wm2, const float* __restrict__ bm2,
                const float* __restrict__ wm3, const float* __restrict__ bm3,
                const int* __restrict__ contrib, float* __restrict__ out, int nN) {
  __shared__ _Float16 A1[WAVES][16][H];
  __shared__ _Float16 A2[WAVES][16][H];

  int lane = threadIdx.x & 31, wv = threadIdx.x >> 5;
  int tile = blockIdx.x * WAVES + wv;
  int n0 = tile * 16;
  if (n0 >= nN) return;
  int mrow = lane & 15, hs = lane >> 4;
  int node = n0 + mrow; if (node >= nN) node = nN - 1;

  {
    const float4* sp = (const float4*)(h + (size_t)node * H + hs * 64);
    v4h* ap = (v4h*)(&A1[wv][mrow][hs * 64]);
#pragma unroll
    for (int j = 0; j < 16; ++j) {
      float4 v = sp[j];
      v4h o; o[0] = (_Float16)v.x; o[1] = (_Float16)v.y; o[2] = (_Float16)v.z; o[3] = (_Float16)v.w;
      ap[j] = o;
    }
  }
  asm volatile("s_wait_dscnt 0" ::: "memory");

  for (int nt = 0; nt < 8; ++nt) {
    int n = nt * 16 + (lane & 15);
    v8f acc;
#pragma unroll
    for (int v = 0; v < 8; ++v) acc[v] = bm1[n];
#pragma unroll
    for (int kb = 0; kb < H; kb += 32) {
      v16h a = a_frag_lds(&A1[wv][0][0], H, kb, lane);
      v16h b = b_frag_h(Wm1, H, nt * 16, kb, lane);
      acc = WMMA_F16(acc, a, b);
    }
#pragma unroll
    for (int v = 0; v < 8; ++v) A2[wv][v + 8 * hs][n] = (_Float16)silu_f(acc[v]);
  }
  asm volatile("s_wait_dscnt 0" ::: "memory");

  float part[8];
#pragma unroll
  for (int v = 0; v < 8; ++v) part[v] = 0.0f;
  for (int nt = 0; nt < 8; ++nt) {
    int n = nt * 16 + (lane & 15);
    float w3 = wm3[n];
    v8f acc;
#pragma unroll
    for (int v = 0; v < 8; ++v) acc[v] = bm2[n];
#pragma unroll
    for (int kb = 0; kb < H; kb += 32) {
      v16h a = a_frag_lds(&A2[wv][0][0], H, kb, lane);
      v16h b = b_frag_h(Wm2, H, nt * 16, kb, lane);
      acc = WMMA_F16(acc, a, b);
    }
#pragma unroll
    for (int v = 0; v < 8; ++v) part[v] += silu_f(acc[v]) * w3;
  }
#pragma unroll
  for (int off = 1; off < 16; off <<= 1) {
#pragma unroll
    for (int v = 0; v < 8; ++v) part[v] += __shfl_xor(part[v], off, 32);
  }
  float b3 = bm3[0];
  if ((lane & 15) == 0) {
#pragma unroll
    for (int v = 0; v < 8; ++v) {
      int nv = n0 + v + 8 * hs;
      if (nv < nN && contrib[nv] == 0) atomicAdd(out, part[v] + b3);
    }
  }
}

// ---------------------------------------------------------------------------
// Launch
// ---------------------------------------------------------------------------
extern "C" void kernel_launch(void* const* d_in, const int* in_sizes, int n_in,
                              void* d_out, int out_size, void* d_ws, size_t ws_size,
                              hipStream_t stream) {
  const float* x    = (const float*)d_in[0];
  const float* r_in = (const float*)d_in[1];
  const int* ei[3]  = { (const int*)d_in[2], (const int*)d_in[3], (const int*)d_in[4] };
  const int* contrib = (const int*)d_in[5];
  const float* emb_w = (const float*)d_in[6];
  const float* emb_b = (const float*)d_in[7];

  int nN = in_sizes[0];
  int nE = in_sizes[2] / 2;

  struct LayerP {
    const float *we1, *be1, *we2, *be2;  // phi_e
    const float *wh1, *bh1, *wh2, *bh2;  // phi_h
    const float *wr1, *br1, *wr2, *br2;  // phi_r
  } L[3];
  for (int l = 0; l < 3; ++l) {
    int b = 8 + l * 12;
    L[l].we1 = (const float*)d_in[b + 0];  L[l].be1 = (const float*)d_in[b + 1];
    L[l].we2 = (const float*)d_in[b + 2];  L[l].be2 = (const float*)d_in[b + 3];
    L[l].wh1 = (const float*)d_in[b + 4];  L[l].bh1 = (const float*)d_in[b + 5];
    L[l].wh2 = (const float*)d_in[b + 6];  L[l].bh2 = (const float*)d_in[b + 7];
    L[l].wr1 = (const float*)d_in[b + 8];  L[l].br1 = (const float*)d_in[b + 9];
    L[l].wr2 = (const float*)d_in[b + 10]; L[l].br2 = (const float*)d_in[b + 11];
  }
  const float* Wm1 = (const float*)d_in[44]; const float* bm1 = (const float*)d_in[45];
  const float* Wm2 = (const float*)d_in[46]; const float* bm2 = (const float*)d_in[47];
  const float* wm3 = (const float*)d_in[48]; const float* bm3 = (const float*)d_in[49];

  // workspace carve-up
  size_t szH = (size_t)nN * H * sizeof(float);
  size_t szR = (size_t)nN * 3 * sizeof(float);
  char* ws = (char*)d_ws;
  float* h_a  = (float*)ws; ws += szH;
  float* h_b  = (float*)ws; ws += szH;
  float* m_b  = (float*)ws; ws += szH;
  float* r_a  = (float*)ws; ws += szR;
  float* r_b  = (float*)ws; ws += szR;
  float* racc = (float*)ws; ws += szR;
  unsigned* xmax = (unsigned*)ws; ws += 256;

  // f16 weight mirrors
  const size_t sz256 = (size_t)H * 256 * sizeof(_Float16);  // 64 KB
  const size_t sz128 = (size_t)H * H * sizeof(_Float16);    // 32 KB
  _Float16 *We1h[3], *We2h[3], *Wr1h[3], *Wh1h[3], *Wh2h[3];
  for (int l = 0; l < 3; ++l) {
    We1h[l] = (_Float16*)ws; ws += sz256;
    We2h[l] = (_Float16*)ws; ws += sz128;
    Wr1h[l] = (_Float16*)ws; ws += sz128;
    Wh1h[l] = (_Float16*)ws; ws += sz256;
    Wh2h[l] = (_Float16*)ws; ws += sz128;
  }
  _Float16* Wm1h = (_Float16*)ws; ws += sz128;
  _Float16* Wm2h = (_Float16*)ws; ws += sz128;

  hipMemsetAsync(xmax, 0, sizeof(unsigned), stream);
  hipMemsetAsync(d_out, 0, sizeof(float), stream);

  // one-time weight conversions (tiny next to 320 GFLOP of GEMMs)
  int g256 = (H * 256 + 255) / 256, g128 = (H * H + 255) / 256;
  for (int l = 0; l < 3; ++l) {
    cvt_w_kernel<<<g256, 256, 0, stream>>>(L[l].we1, We1h[l], H, 257, 256);
    cvt_w_kernel<<<g128, 256, 0, stream>>>(L[l].we2, We2h[l], H, H, H);
    cvt_w_kernel<<<g128, 256, 0, stream>>>(L[l].wr1, Wr1h[l], H, H, H);
    cvt_w_kernel<<<g256, 256, 0, stream>>>(L[l].wh1, Wh1h[l], H, 256, 256);
    cvt_w_kernel<<<g128, 256, 0, stream>>>(L[l].wh2, Wh2h[l], H, H, H);
  }
  cvt_w_kernel<<<g128, 256, 0, stream>>>(Wm1, Wm1h, H, H, H);
  cvt_w_kernel<<<g128, 256, 0, stream>>>(Wm2, Wm2h, H, H, H);

  xmax_kernel<<<(nN + 255) / 256, 256, 0, stream>>>(x, xmax, nN);
  embed_kernel<<<(nN * H + 255) / 256, 256, 0, stream>>>(x, xmax, emb_w, emb_b, h_a, nN);

  int eTiles = (nE + 15) / 16;
  int eBlocks = (eTiles + WAVES - 1) / WAVES;
  int nTiles = (nN + 15) / 16;
  int nBlocks = (nTiles + WAVES - 1) / WAVES;

  // layer order: edge_index2 -> layers[0], edge_index1 -> layers[1], edge_index0 -> layers[2]
  const int* eo[3] = { ei[2], ei[1], ei[0] };
  const float* hin[3]  = { h_a, h_b, h_a };
  float*       hout[3] = { h_b, h_a, h_b };
  const float* rin[3]  = { r_in, r_a, r_b };
  float*       rout[3] = { r_a, r_b, r_a };

  for (int l = 0; l < 3; ++l) {
    hipMemsetAsync(m_b, 0, szH, stream);
    hipMemsetAsync(racc, 0, szR, stream);
    egcl_edge_kernel<<<eBlocks, TPB, 0, stream>>>(
        hin[l], rin[l], eo[l],
        L[l].we1, We1h[l], L[l].be1, We2h[l], L[l].be2,
        Wr1h[l], L[l].br1, L[l].wr2, L[l].br2,
        m_b, racc, nE);
    egcl_node_kernel<<<nBlocks, TPB, 0, stream>>>(
        hin[l], m_b, Wh1h[l], L[l].bh1, Wh2h[l], L[l].bh2, hout[l], nN);
    radd_kernel<<<(nN * 3 + 255) / 256, 256, 0, stream>>>(rin[l], racc, rout[l], nN * 3);
  }

  mlp_kernel<<<nBlocks, TPB, 0, stream>>>(
      h_b, Wm1h, bm1, Wm2h, bm2, wm3, bm3, contrib, (float*)d_out, nN);
}
